// SAINT_74148315398472
// MI455X (gfx1250) — matrix operand order repaired and checked by reading, not compiled, not run on hardware
//
#include <hip/hip_runtime.h>

#define N_NODES 50000
#define N_EDGES 800000
#define IN_C 96
#define HID_C 96
#define OUT_C 64

typedef __attribute__((ext_vector_type(16))) __bf16 v16bf;
typedef __attribute__((ext_vector_type(8)))  float  v8f;

// ---------------------------------------------------------------------------
// WMMA bf16 16x16x32 operand layouts used by this code (contiguous k order, not the ISA table order):
//  A (16x32, MxK): lane L holds M=L%16; element i (0..15):
//      half = L/16
//      K = half*16 + i                        (pairs i,i+1 are K,K+1)
//  B (32x16, KxN): lane L holds N=L%16; element i: K = (L/16)*16 + i
//  C (16x16 f32): lane L holds N=L%16; VGPR j -> M = j + (L/16)*8
// ---------------------------------------------------------------------------
__device__ __forceinline__ int kpatA(int lane, int i) {
    int half = (lane >> 4) & 1;
    int v = i >> 1;
    return (half << 4) + i;
}

// Build an A fragment from an fp32 row-major matrix with row stride 96.
__device__ __forceinline__ v16bf load_a_frag96(const float* __restrict__ M,
                                               int row, int k0, int lane) {
    v16bf a;
    const float* base = M + (size_t)row * 96 + k0;
#pragma unroll
    for (int i = 0; i < 16; i += 2) {
        int k = kpatA(lane, i);
        float2 f = *(const float2*)(base + k);
        a[i]     = (__bf16)f.x;
        a[i + 1] = (__bf16)f.y;
    }
    return a;
}

// ---------------------------------------------------------------------------
// Repack fp32 weight W[outC x K] (row-major) into bf16 B-fragments:
// pack[((kt*NT + nt)*32 + lane)*16 + i] = bf16( W[(nt*16 + lane%16), kt*32 + (lane/16)*16 + i] )
// ---------------------------------------------------------------------------
__global__ void pack_weight_kernel(const float* __restrict__ W,
                                   __bf16* __restrict__ pack,
                                   int K, int NT, int KT) {
    int t = blockIdx.x * blockDim.x + threadIdx.x;
    int total = KT * NT * 32;
    if (t >= total) return;
    int lane = t & 31;
    int nt = (t >> 5) % NT;
    int kt = (t >> 5) / NT;
    int n = nt * 16 + (lane & 15);
    int kbase = kt * 32 + ((lane >> 4) << 4);
    __bf16* dst = pack + ((size_t)(kt * NT + nt) * 32 + lane) * 16;
    const float* src = W + (size_t)n * K + kbase;
#pragma unroll
    for (int i = 0; i < 16; ++i) dst[i] = (__bf16)src[i];
}

__global__ void zero_kernel(float* __restrict__ p, int n) {
    int t = blockIdx.x * blockDim.x + threadIdx.x;
    if (t < n) p[t] = 0.0f;
}

// agg[dst] += x[src], fp32 atomics; one thread per (edge, float4 chunk of 96ch)
__global__ void scatter_add_kernel(const float* __restrict__ x,
                                   const long long* __restrict__ ei,
                                   float* __restrict__ agg) {
    int t = blockIdx.x * blockDim.x + threadIdx.x;
    int e = t / 24;
    int c = t - e * 24;
    if (e >= N_EDGES) return;
    long long s = ei[e];
    long long d = ei[N_EDGES + e];
    const float4 v = *(const float4*)(x + (size_t)s * 96 + c * 4);
    float* o = agg + (size_t)d * 96 + c * 4;
    atomicAdd(o + 0, v.x);
    atomicAdd(o + 1, v.y);
    atomicAdd(o + 2, v.z);
    atomicAdd(o + 3, v.w);
}

// xout = relu(agg @ Wrel^T + xin @ Wroot^T + b); per wave: 16 rows x 96 cols
__global__ void __launch_bounds__(128)
gconv_gemm_kernel(const float* __restrict__ agg, const float* __restrict__ xin,
                  const __bf16* __restrict__ packRel, const __bf16* __restrict__ packRoot,
                  const float* __restrict__ bias, float* __restrict__ xout) {
    const int lane = threadIdx.x & 31;
    const int wave = threadIdx.x >> 5;
    const int rt = blockIdx.x * 4 + wave;
    const int RT = (N_NODES + 15) / 16;   // 3125 (exact: 3125*16 == 50000)
    if (rt >= RT) return;                  // uniform per wave -> EXEC stays all-ones
    const int rbase = rt * 16;
    const int arow = rbase + (lane & 15);

    v8f c[6] = {};

#pragma unroll
    for (int kt = 0; kt < 3; ++kt) {
        v16bf a = load_a_frag96(agg, arow, kt * 32, lane);
#pragma unroll
        for (int nt = 0; nt < 6; ++nt) {
            v16bf b = *(const v16bf*)(packRel + ((size_t)(kt * 6 + nt) * 32 + lane) * 16);
            c[nt] = __builtin_amdgcn_wmma_f32_16x16x32_bf16(
                false, a, false, b, (short)0, c[nt], false, false);
        }
    }
#pragma unroll
    for (int kt = 0; kt < 3; ++kt) {
        v16bf a = load_a_frag96(xin, arow, kt * 32, lane);
#pragma unroll
        for (int nt = 0; nt < 6; ++nt) {
            v16bf b = *(const v16bf*)(packRoot + ((size_t)(kt * 6 + nt) * 32 + lane) * 16);
            c[nt] = __builtin_amdgcn_wmma_f32_16x16x32_bf16(
                false, a, false, b, (short)0, c[nt], false, false);
        }
    }

    const int ncol = lane & 15;
    const int mOff = (lane >> 4) * 8;
#pragma unroll
    for (int nt = 0; nt < 6; ++nt) {
        int col = nt * 16 + ncol;
        float bv = bias[col];
#pragma unroll
        for (int j = 0; j < 8; ++j) {
            int row = rbase + mOff + j;
            float v = c[nt][j] + bv;
            v = v > 0.0f ? v : 0.0f;
            xout[(size_t)row * 96 + col] = v;
        }
    }
}

// logits = [x1|x2|x3] @ Wlin^T + b; K=288 as 3 segments of 96, N=64
__global__ void __launch_bounds__(128)
final_gemm_kernel(const float* __restrict__ x1, const float* __restrict__ x2,
                  const float* __restrict__ x3, const __bf16* __restrict__ packLin,
                  const float* __restrict__ bias, float* __restrict__ logits) {
    const int lane = threadIdx.x & 31;
    const int wave = threadIdx.x >> 5;
    const int rt = blockIdx.x * 4 + wave;
    const int RT = (N_NODES + 15) / 16;
    if (rt >= RT) return;
    const int rbase = rt * 16;
    const int arow = rbase + (lane & 15);

    v8f c[4] = {};
    const float* xs[3] = {x1, x2, x3};

#pragma unroll
    for (int seg = 0; seg < 3; ++seg) {
#pragma unroll
        for (int kk = 0; kk < 3; ++kk) {
            v16bf a = load_a_frag96(xs[seg], arow, kk * 32, lane);
            int kt = seg * 3 + kk;
#pragma unroll
            for (int nt = 0; nt < 4; ++nt) {
                v16bf b = *(const v16bf*)(packLin + ((size_t)(kt * 4 + nt) * 32 + lane) * 16);
                c[nt] = __builtin_amdgcn_wmma_f32_16x16x32_bf16(
                    false, a, false, b, (short)0, c[nt], false, false);
            }
        }
    }

    const int ncol = lane & 15;
    const int mOff = (lane >> 4) * 8;
#pragma unroll
    for (int nt = 0; nt < 4; ++nt) {
        int col = nt * 16 + ncol;
        float bv = bias[col];
#pragma unroll
        for (int j = 0; j < 8; ++j) {
            int row = rbase + mOff + j;
            logits[(size_t)row * 64 + col] = c[nt][j] + bv;
        }
    }
}

// One wave32 per row; 64 classes -> 2 values per lane; shfl_xor reductions.
__global__ void __launch_bounds__(256)
log_softmax_kernel(const float* __restrict__ logits, float* __restrict__ out) {
    int lane = threadIdx.x & 31;
    int row = (blockIdx.x * blockDim.x + threadIdx.x) >> 5;
    if (row >= N_NODES) return;
    const float* p = logits + (size_t)row * 64;
    float a = p[lane];
    float b = p[lane + 32];
    float m = fmaxf(a, b);
#pragma unroll
    for (int off = 16; off >= 1; off >>= 1)
        m = fmaxf(m, __shfl_xor(m, off, 32));
    float s = __expf(a - m) + __expf(b - m);
#pragma unroll
    for (int off = 16; off >= 1; off >>= 1)
        s += __shfl_xor(s, off, 32);
    float lse = m + __logf(s);
    float* o = out + (size_t)row * 64;
    o[lane]      = a - lse;
    o[lane + 32] = b - lse;
}

extern "C" void kernel_launch(void* const* d_in, const int* in_sizes, int n_in,
                              void* d_out, int out_size, void* d_ws, size_t ws_size,
                              hipStream_t stream) {
    (void)in_sizes; (void)n_in; (void)out_size; (void)ws_size;

    const float*     x0      = (const float*)d_in[0];
    const long long* ei      = (const long long*)d_in[1];
    const float*     W1_rel  = (const float*)d_in[2];
    const float*     b1      = (const float*)d_in[3];
    const float*     W1_root = (const float*)d_in[4];
    const float*     W2_rel  = (const float*)d_in[5];
    const float*     b2      = (const float*)d_in[6];
    const float*     W2_root = (const float*)d_in[7];
    const float*     W3_rel  = (const float*)d_in[8];
    const float*     b3      = (const float*)d_in[9];
    const float*     W3_root = (const float*)d_in[10];
    const float*     W_lin   = (const float*)d_in[11];
    const float*     b_lin   = (const float*)d_in[12];
    float*           out     = (float*)d_out;

    char* ws = (char*)d_ws;
    size_t off = 0;
    auto alloc = [&](size_t bytes) -> void* {
        void* p = ws + off;
        off += (bytes + 255) & ~(size_t)255;
        return p;
    };

    const size_t featBytes = (size_t)N_NODES * 96 * sizeof(float);
    float* agg = (float*)alloc(featBytes);   // reused as logits (N x 64 fits)
    float* x1  = (float*)alloc(featBytes);
    float* x2  = (float*)alloc(featBytes);
    float* x3  = (float*)alloc(featBytes);

    const size_t pack96Bytes = (size_t)3 * 6 * 32 * 16 * sizeof(__bf16);   // KT=3,NT=6
    const size_t packLinBytes = (size_t)9 * 4 * 32 * 16 * sizeof(__bf16);  // KT=9,NT=4
    __bf16* p1r = (__bf16*)alloc(pack96Bytes);
    __bf16* p1o = (__bf16*)alloc(pack96Bytes);
    __bf16* p2r = (__bf16*)alloc(pack96Bytes);
    __bf16* p2o = (__bf16*)alloc(pack96Bytes);
    __bf16* p3r = (__bf16*)alloc(pack96Bytes);
    __bf16* p3o = (__bf16*)alloc(pack96Bytes);
    __bf16* plin = (__bf16*)alloc(packLinBytes);

    // Pack all weights to bf16 WMMA B-fragments (tiny kernels).
    const int pt96 = 3 * 6 * 32, ptLin = 9 * 4 * 32;
    pack_weight_kernel<<<(pt96 + 255) / 256, 256, 0, stream>>>(W1_rel,  p1r, 96, 6, 3);
    pack_weight_kernel<<<(pt96 + 255) / 256, 256, 0, stream>>>(W1_root, p1o, 96, 6, 3);
    pack_weight_kernel<<<(pt96 + 255) / 256, 256, 0, stream>>>(W2_rel,  p2r, 96, 6, 3);
    pack_weight_kernel<<<(pt96 + 255) / 256, 256, 0, stream>>>(W2_root, p2o, 96, 6, 3);
    pack_weight_kernel<<<(pt96 + 255) / 256, 256, 0, stream>>>(W3_rel,  p3r, 96, 6, 3);
    pack_weight_kernel<<<(pt96 + 255) / 256, 256, 0, stream>>>(W3_root, p3o, 96, 6, 3);
    pack_weight_kernel<<<(ptLin + 255) / 256, 256, 0, stream>>>(W_lin, plin, 288, 4, 9);

    const int zeroN = N_NODES * 96;
    const int zeroGrid = (zeroN + 255) / 256;
    const long long scatterThreads = (long long)N_EDGES * 24;
    const int scatterGrid = (int)((scatterThreads + 255) / 256);
    const int gemmGrid = ((N_NODES + 15) / 16 + 3) / 4;   // 4 waves/block

    auto layer = [&](const float* xin, const __bf16* pRel, const __bf16* pRoot,
                     const float* b, float* xout) {
        zero_kernel<<<zeroGrid, 256, 0, stream>>>(agg, zeroN);
        scatter_add_kernel<<<scatterGrid, 256, 0, stream>>>(xin, ei, agg);
        gconv_gemm_kernel<<<gemmGrid, 128, 0, stream>>>(agg, xin, pRel, pRoot, b, xout);
    };

    layer(x0, p1r, p1o, b1, x1);
    layer(x1, p2r, p2o, b2, x2);
    layer(x2, p3r, p3o, b3, x3);

    float* logits = agg;  // reuse
    final_gemm_kernel<<<gemmGrid, 128, 0, stream>>>(x1, x2, x3, plin, b_lin, logits);

    const int smGrid = (int)(((long long)N_NODES * 32 + 255) / 256);
    log_softmax_kernel<<<smGrid, 256, 0, stream>>>(logits, out);
}
